// DynamicSnakeConv_54013508715029
// MI455X (gfx1250) — matrix operand order repaired and checked by reading, not compiled, not run on hardware
//
#include <hip/hip_runtime.h>

// DynamicSnakeConv for MI455X (gfx1250, wave32).
// Pipeline: pack_conv_w -> extract_col -> offset_conv -> snake_main (WMMA f16 GEMM).
// Workspace layout (needs ~6.6 MB):
//   [0)          y_new  : 8*3*256*256 f32  = 6,291,456 B
//   [6291456)    colx   : 8*256*64   f16  =   262,144 B   (x[:,:,:,0], [n][y][ci])
//   [6553600)    wpack  : 8*6*32*16  f16  =    49,152 B   (A fragments, lane layout)

#define N_   8
#define CIN  64
#define COUT 128
#define H_   256
#define W_   256
#define KS_  3

typedef __attribute__((ext_vector_type(16))) _Float16 v16h;
typedef __attribute__((ext_vector_type(8)))  _Float16 v8h;
typedef __attribute__((ext_vector_type(8)))  float    v8f;

union V16HU { v16h v; v8h h[2]; };

__device__ __forceinline__ float fast_tanh(float x) {
  // tanh(x) = 1 - 2/(exp(2x)+1); saturates correctly at +-1 for large |x|.
  float e = __expf(2.0f * x);
  return 1.0f - 2.0f / (e + 1.0f);
}

// ---------------------------------------------------------------------------
// Pack conv_w [128][64][3][1] f32 into WMMA A-fragment order, f16.
// K-index = t*64 + ci.  Per ISA 16-bit A 16x32 layout:
//   lane<16 : row M=lane,    halfs j=0..7 -> K=kc*32+j,    j=8..15 -> K=kc*32+16+(j-8)
//   lane>=16: row M=lane-16, halfs j=0..7 -> K=kc*32+8+j,  j=8..15 -> K=kc*32+24+(j-8)
// wpack[((cotile*6+kc)*32+lane)*16 + j]
// ---------------------------------------------------------------------------
__global__ void pack_conv_w(const float* __restrict__ conv_w,
                            _Float16* __restrict__ wpack) {
  int tid = blockIdx.x * blockDim.x + threadIdx.x;
  if (tid >= 8 * 6 * 32) return;
  int lane   = tid & 31;
  int kc     = (tid >> 5) % 6;
  int cotile = tid / (32 * 6);
  int row = lane & 15, hi = lane >> 4;
  int co = cotile * 16 + row;
  _Float16 frag[16];
#pragma unroll
  for (int jj = 0; jj < 16; ++jj) {
    int K  = kc * 32 + (jj >> 3) * 16 + hi * 8 + (jj & 7);  // 0..191
    int t  = K / 64;
    int ci = K % 64;
    frag[jj] = (_Float16)conv_w[(co * CIN + ci) * KS_ + t];
  }
#pragma unroll
  for (int jj = 0; jj < 16; ++jj) wpack[tid * 16 + jj] = frag[jj];
}

// ---------------------------------------------------------------------------
// colx[n][y][ci] = (f16) x[n][ci][y][0]   (gx == -1 => only column 0 sampled)
// ---------------------------------------------------------------------------
__global__ void extract_col(const float* __restrict__ x,
                            _Float16* __restrict__ colx) {
  int idx = blockIdx.x * blockDim.x + threadIdx.x;
  if (idx >= N_ * H_ * CIN) return;
  int ci = idx & 63;
  int y  = (idx >> 6) & 255;
  int n  = idx >> 14;
  colx[idx] = (_Float16)x[(((size_t)(n * CIN + ci)) * H_ + y) * W_];
}

// ---------------------------------------------------------------------------
// Offset conv (3x3, 64->3ch, pad 1) + tanh + cumsum combine.
// y_new[n][0] = t0+t1 ; y_new[n][1] = t1 ; y_new[n][2] = t1+t2
// Block: 256 thr = 16(h) x 16(w-groups), 4 px/thread along w. Tile 16x64.
// ---------------------------------------------------------------------------
__global__ __launch_bounds__(256) void offset_conv(const float* __restrict__ x,
                                                   const float* __restrict__ offset_w,
                                                   float* __restrict__ ynew) {
  __shared__ float wsm[3 * CIN * 9];  // first 3 out-channels, flat == offset_w[:1728]
  __shared__ float xs[18 * 66];       // (16+2) rows x (64+2) cols halo tile
  int b   = blockIdx.x;
  int wti = b & 3, hti = (b >> 2) & 15, n = b >> 6;
  int y0 = hti * 16, w0 = wti * 64;
  int tid  = threadIdx.x;
  int lrow = tid >> 4;
  int lcol = (tid & 15) * 4;

  for (int i = tid; i < 3 * CIN * 9; i += 256) wsm[i] = offset_w[i];

  float acc0[4] = {0, 0, 0, 0}, acc1[4] = {0, 0, 0, 0}, acc2[4] = {0, 0, 0, 0};
  for (int ci = 0; ci < CIN; ++ci) {
    __syncthreads();  // protects wsm (ci==0) and xs reuse (ci>0)
    const float* xp = x + ((size_t)(n * CIN + ci)) * H_ * W_;
    for (int i = tid; i < 18 * 66; i += 256) {
      int r = i / 66, c = i % 66;
      int gy = y0 - 1 + r, gx = w0 - 1 + c;
      float v = 0.0f;
      if (gy >= 0 && gy < H_ && gx >= 0 && gx < W_) v = xp[gy * W_ + gx];
      xs[i] = v;
    }
    __syncthreads();
    const float* wb = &wsm[ci * 9];
#pragma unroll
    for (int dy = 0; dy < 3; ++dy) {
      float xv[6];
#pragma unroll
      for (int q = 0; q < 6; ++q) xv[q] = xs[(lrow + dy) * 66 + lcol + q];
#pragma unroll
      for (int dx = 0; dx < 3; ++dx) {
        float g0 = wb[0 * CIN * 9 + dy * 3 + dx];
        float g1 = wb[1 * CIN * 9 + dy * 3 + dx];
        float g2 = wb[2 * CIN * 9 + dy * 3 + dx];
#pragma unroll
        for (int p = 0; p < 4; ++p) {
          float xvv = xv[dx + p];
          acc0[p] = fmaf(xvv, g0, acc0[p]);
          acc1[p] = fmaf(xvv, g1, acc1[p]);
          acc2[p] = fmaf(xvv, g2, acc2[p]);
        }
      }
    }
  }
  int y = y0 + lrow;
  size_t plane = (size_t)H_ * W_;
  size_t baseo = (size_t)n * 3 * plane + (size_t)y * W_ + (w0 + lcol);
#pragma unroll
  for (int p = 0; p < 4; ++p) {
    float t0 = fast_tanh(acc0[p]);
    float t1 = fast_tanh(acc1[p]);
    float t2 = fast_tanh(acc2[p]);
    ynew[baseo + p]             = t0 + t1;
    ynew[baseo + plane + p]     = t1;
    ynew[baseo + 2 * plane + p] = t1 + t2;
  }
}

// ---------------------------------------------------------------------------
// Fused degenerate grid-sample + GEMM (WMMA f16 -> f32).
// Block = (n, j, 64 output columns). Build B panel samp[w'][K=192] f16 in LDS,
// then 8 waves x (16 co) x 4 (16 w') tiles, K = 6 chunks of 32.
// Restructured: A-frag global loads issued at entry (latency hidden behind
// panel build + barrier); per K-chunk, all 8 ds_load_b128 hoisted ahead of the
// 4 back-to-back WMMAs so a single dscnt wait covers the whole clause.
// ---------------------------------------------------------------------------
__global__ __launch_bounds__(256) void snake_main(const _Float16* __restrict__ colx,
                                                  const float* __restrict__ ynewg,
                                                  const _Float16* __restrict__ wpack,
                                                  float* __restrict__ out) {
  // column stride 200 halfs (400B = 100 dwords): conflict-free b128 column reads
  __shared__ __attribute__((aligned(16))) _Float16 samp[64 * 200];  // 25,600 B

  int b   = blockIdx.x;
  int wti = b & 3;
  int j   = (b >> 2) & 255;
  int n   = b >> 10;
  int wbase = wti * 64;
  int tid = threadIdx.x;
  int lane   = tid & 31;
  int wv     = tid >> 5;      // co tile 0..7
  int colsel = lane & 15;
  int hi     = lane >> 4;
  const _Float16* colp = colx + (size_t)n * H_ * CIN;

  // ---- issue A-fragment loads first: latency overlaps the panel build ----
  V16HU afrag[6];
#pragma unroll
  for (int kc = 0; kc < 6; ++kc)
    afrag[kc].v = *(const v16h*)(wpack + (((size_t)wv * 6 + kc) * 32 + lane) * 16);

  // ---- build sampled panel: 64 w' x 3 t x 8 ci-octets = 1536 groups ----
#pragma unroll
  for (int m = 0; m < 6; ++m) {
    int g    = tid + m * 256;    // 0..1535
    int ci8  = g & 7;
    int pair = g >> 3;           // 0..191
    int t    = pair % 3;
    int wp   = pair / 3;         // 0..63
    int r = 3 * j + t;           // row in the ks*h reshape
    int k = r >> 8;              // which y_new plane
    int y = r & 255;
    int wg = wbase + wp;
    float yn   = ynewg[(((size_t)n * 3 + k) * H_ + y) * W_ + wg];
    float base = (float)((k + y + wg) % 3 - 1);      // vals[(flat)%3] quirk
    float iy   = (base + yn) * (256.0f / 255.0f) - 0.5f;
    float fy   = floorf(iy);
    int   y0i  = (int)fy;
    float wy1  = iy - fy, wy0 = 1.0f - wy1;
    v8h v0 = {}, v1 = {};
    if (y0i >= 0 && y0i < H_)         v0 = *(const v8h*)(colp + y0i * CIN + ci8 * 8);
    int y1i = y0i + 1;
    if (y1i >= 0 && y1i < H_)         v1 = *(const v8h*)(colp + y1i * CIN + ci8 * 8);
    v8h res;
#pragma unroll
    for (int q = 0; q < 8; ++q)
      res[q] = (_Float16)(0.5f * (wy0 * (float)v0[q] + wy1 * (float)v1[q]));
    *(v8h*)(&samp[wp * 200 + t * 64 + ci8 * 8]) = res;
  }
  __syncthreads();

  // ---- WMMA phase: all lanes active (EXEC all-ones) ----
  int cobase = wv * 16;
  v8f acc[4] = {{}, {}, {}, {}};
#pragma unroll
  for (int kc = 0; kc < 6; ++kc) {
    V16HU bfrag[4];
#pragma unroll
    for (int wt = 0; wt < 4; ++wt) {
      // B 32x16 f16: lane<16 holds col N=lane K=0..7,16..23; lane>=16 holds K=8..15,24..31
      const _Float16* bp = &samp[(wt * 16 + colsel) * 200 + kc * 32 + hi * 8];
      bfrag[wt].h[0] = *(const v8h*)(bp);
      bfrag[wt].h[1] = *(const v8h*)(bp + 16);
    }
#pragma unroll
    for (int wt = 0; wt < 4; ++wt)
      acc[wt] = __builtin_amdgcn_wmma_f32_16x16x32_f16(false, afrag[kc].v,
                                                       false, bfrag[wt].v,
                                                       (short)0, acc[wt],
                                                       false, false);
  }

#pragma unroll
  for (int wt = 0; wt < 4; ++wt) {
    int wg = wbase + wt * 16 + colsel;
#pragma unroll
    for (int i = 0; i < 8; ++i) {
      int co = cobase + hi * 8 + i;  // C/D layout: VGPR i -> M=i(+8 for hi lanes)
      out[(((size_t)n * COUT + co) * H_ + j) * W_ + wg] = acc[wt][i];
    }
  }
}

// ---------------------------------------------------------------------------
extern "C" void kernel_launch(void* const* d_in, const int* in_sizes, int n_in,
                              void* d_out, int out_size, void* d_ws, size_t ws_size,
                              hipStream_t stream) {
  (void)in_sizes; (void)n_in; (void)out_size; (void)ws_size;
  const float* x        = (const float*)d_in[0];
  const float* offset_w = (const float*)d_in[1];
  const float* conv_w   = (const float*)d_in[2];
  float* out = (float*)d_out;

  char* ws = (char*)d_ws;
  float*    ynewg = (float*)ws;                   // 6,291,456 B
  _Float16* colx  = (_Float16*)(ws + 6291456);    //   262,144 B
  _Float16* wpack = (_Float16*)(ws + 6553600);    //    49,152 B

  pack_conv_w<<<6, 256, 0, stream>>>(conv_w, wpack);
  extract_col<<<(N_ * H_ * CIN + 255) / 256, 256, 0, stream>>>(x, colx);
  offset_conv<<<N_ * 16 * 4, 256, 0, stream>>>(x, offset_w, ynewg);
  snake_main<<<N_ * 256 * 4, 256, 0, stream>>>(colx, ynewg, wpack, out);
}
